// RNN_69836168233531
// MI455X (gfx1250) — matrix operand order repaired and checked by reading, not compiled, not run on hardware
//
#include <hip/hip_runtime.h>
#include <hip/hip_bf16.h>
#include <math.h>

// Problem sizes (fixed by the reference)
#define NB 64
#define TT 512
#define DD 512
#define HH 512

typedef __attribute__((ext_vector_type(16))) __bf16          v16bf;
typedef __attribute__((ext_vector_type(8)))  float           v8f;
typedef __attribute__((ext_vector_type(8)))  unsigned short  v8us;
typedef __attribute__((ext_vector_type(16))) unsigned short  v16us;

// Async-copy pointer types: builtin expects v4i addrspace(1)* / addrspace(3)*
// (diagnostic showed param0 as "__vector_size__(4*sizeof(int)) int __device__ *")
typedef int v4i __attribute__((vector_size(16)));
typedef v4i __attribute__((address_space(1)))* as1_v4i;
typedef v4i __attribute__((address_space(3)))* as3_v4i;

#if __has_builtin(__builtin_amdgcn_global_load_async_to_lds_b128)
#define HAVE_ASYNC_LDS 1
#else
#define HAVE_ASYNC_LDS 0
#endif

// f32 -> bf16 via hardware convert (RNE), bit pattern as ushort
__device__ __forceinline__ unsigned short f2bfu(float f) {
    return __builtin_bit_cast(unsigned short, (__bf16)f);
}

__device__ __forceinline__ float fast_tanh(float x) {
#if __has_builtin(__builtin_amdgcn_tanhf)
    return __builtin_amdgcn_tanhf(x);
#elif __has_builtin(__builtin_amdgcn_tanh_f32)
    return __builtin_amdgcn_tanh_f32(x);
#else
    return tanhf(x);
#endif
}

__device__ __forceinline__ void wait_async_le8() {
#if HAVE_ASYNC_LDS
#if __has_builtin(__builtin_amdgcn_s_wait_asynccnt)
    __builtin_amdgcn_s_wait_asynccnt(8);
#else
    asm volatile("s_wait_asynccnt 0x8" ::: "memory");
#endif
#endif
}
__device__ __forceinline__ void wait_async_0() {
#if HAVE_ASYNC_LDS
#if __has_builtin(__builtin_amdgcn_s_wait_asynccnt)
    __builtin_amdgcn_s_wait_asynccnt(0);
#else
    asm volatile("s_wait_asynccnt 0x0" ::: "memory");
#endif
#endif
}

// -----------------------------------------------------------------------------
// Kernel 0: swizzle a 512x512 f32 weight matrix (K rows x N cols, row-major)
// into bf16 B-fragment order. Fragment (nt, kt) covers B[kt*32..+32, nt*16..+16].
// Per CDNA5 ISA (CDNA5 ISA, 16-bit B 32x16): lane c<16 holds K={0..7,16..23} of col c,
// lane c>=16 holds K={8..15,24..31} of col c-16 (same table order as A).
// Dest index: ((nt*16 + kt)*32 + c)*16 + h  -> each lane's fragment is one
// contiguous 32-byte chunk => 2x global_load_b128 per fragment at GEMM time.
// -----------------------------------------------------------------------------
__global__ void __launch_bounds__(256) swizzle_w(const float* __restrict__ W,
                                                 unsigned short* __restrict__ Ws) {
    int i  = blockIdx.x * 256 + threadIdx.x;   // 0 .. 262143
    int h  = i & 15;
    int c  = (i >> 4) & 31;
    int kt = (i >> 9) & 15;
    int nt = i >> 13;
    int row = kt * 32 + ((c & 16) >> 1) + (h & 7) + ((h & 8) << 1);   // K index
    int col = nt * 16 + (c & 15);              // N index
    Ws[i] = f2bfu(W[row * HH + col]);
}

// -----------------------------------------------------------------------------
// Kernel 1: xW = x @ Wx + b, written f32 into d_out (N*T, H).
// Grid (256, 8), 256 threads = 8 waves. Each wave owns a 16x64 strip:
// A (16x32 bf16) converted on the fly from f32 x; B fragments from swizzled Wxs.
// -----------------------------------------------------------------------------
__global__ void __launch_bounds__(256) gemm_xw(const float* __restrict__ x,
                                               const unsigned short* __restrict__ Wxs,
                                               const float* __restrict__ b,
                                               float* __restrict__ out) {
    const int lane = threadIdx.x & 31;
    const int w    = threadIdx.x >> 5;
    const int row0 = blockIdx.x * 128 + w * 16;   // 32768 rows total
    const int col0 = blockIdx.y * 64;
    const int m    = lane & 15;
    const int kb   = (lane & 16) >> 1;            // 0 or 8 (A-layout half select)
    const float* arow = x + (long)(row0 + m) * DD;

    v8f acc[4] = {};                              // 4 N-tiles of 16x16 f32

    for (int kt = 0; kt < 16; ++kt) {
        // A fragment: lane<16 -> K {kb..kb+7, 16+kb..}, lane>=16 -> kb=8 halves
        v16bf a;
        #pragma unroll
        for (int p = 0; p < 8; ++p) {
            a[p]     = (__bf16)arow[kt * 32 + kb + p];
            a[8 + p] = (__bf16)arow[kt * 32 + 16 + kb + p];
        }
        #pragma unroll
        for (int j = 0; j < 4; ++j) {
            int nt = (col0 >> 4) + j;
            const unsigned short* bp = Wxs + ((((nt * 16 + kt) * 32) + lane) << 4);
            v16bf bm = __builtin_bit_cast(v16bf, *(const v16us*)bp);
            acc[j] = __builtin_amdgcn_wmma_f32_16x16x32_bf16(
                         false, a, false, bm, (short)0, acc[j], false, false);
        }
    }

    // C/D layout: VGPR r: lanes 0-15 -> M=r, lanes 16-31 -> M=8+r, N=lane&15
    const int mm = (lane >> 4) << 3;
    #pragma unroll
    for (int j = 0; j < 4; ++j) {
        int col = col0 + j * 16 + (lane & 15);
        float bias = b[col];
        float* op = out + (long)(row0 + mm) * HH + col;
        #pragma unroll
        for (int r = 0; r < 8; ++r)
            op[(long)r * HH] = acc[j][r] + bias;
    }
}

// -----------------------------------------------------------------------------
// Kernel 2: persistent recurrence. 4 blocks x 256 threads; block owns 16 batch
// rows for all T steps (rows are independent sequences -> no global sync).
// h lives in LDS as bf16 (A operand); Wh fragments stream from L2 (swizzled).
// xw_t is double-buffered in LDS: the copy of xw(t+1) is issued as
// GLOBAL_LOAD_ASYNC_TO_LDS_B128 (ASYNCcnt) right after step-t WMMAs, hiding
// the 32 KB/step staging behind the GEMM. h_t overwrites xw_t in d_out.
// -----------------------------------------------------------------------------
__global__ void __launch_bounds__(256) rnn_scan(const float* __restrict__ h0,
                                                const unsigned short* __restrict__ Whs,
                                                float* __restrict__ out) {
    __shared__ __align__(16) unsigned short hbuf[16 * 520];     // bf16 h, padded
    __shared__ __align__(16) float          xbuf[2][16 * 516];  // f32 xw_t, padded

    const int tid  = threadIdx.x;
    const int lane = tid & 31;
    const int w    = tid >> 5;
    const int n0   = blockIdx.x * 16;   // batch-row group

    // staging geometry: each thread owns 32 consecutive floats of one row
    const int sr  = tid >> 4;           // 0..15
    const int sc0 = (tid & 15) * 32;

    auto issue_stage = [&](int t, float* buf) {
        const float* src = out + ((long)(n0 + sr) * TT + t) * HH + sc0;
        float* dst = &buf[sr * 516 + sc0];
#if HAVE_ASYNC_LDS
        #pragma unroll
        for (int q = 0; q < 8; ++q)
            __builtin_amdgcn_global_load_async_to_lds_b128(
                (as1_v4i)(src + q * 4), (as3_v4i)(dst + q * 4), 0, 0);
#else
        #pragma unroll
        for (int q = 0; q < 8; ++q)
            ((float4*)dst)[q] = ((const float4*)src)[q];
#endif
    };

    // init h state from h0
    {
        const float* src = h0 + (long)(n0 + sr) * HH + sc0;
        #pragma unroll
        for (int q = 0; q < 32; ++q)
            hbuf[sr * 520 + sc0 + q] = f2bfu(src[q]);
    }
    issue_stage(0, xbuf[0]);            // prefetch xw(0)
    __syncthreads();

    const int m  = lane & 15;
    const int kb = (lane & 16) >> 1;
    const int mm = (lane >> 4) << 3;

    for (int t = 0; t < TT; ++t) {
        const int cur = t & 1;

        // hW = h @ Wh : each wave computes 4 of the 32 N-tiles, K = 512
        v8f acc[4] = {};
        for (int kt = 0; kt < 16; ++kt) {
            const unsigned short* ap = &hbuf[m * 520 + kt * 32 + kb];
            v8us alo = *(const v8us*)ap;          // ds_load_b128
            v8us ahi = *(const v8us*)(ap + 16);   // ds_load_b128
            v16us au = __builtin_shufflevector(alo, ahi,
                         0,1,2,3,4,5,6,7,8,9,10,11,12,13,14,15);
            v16bf a = __builtin_bit_cast(v16bf, au);
            #pragma unroll
            for (int j = 0; j < 4; ++j) {
                int nt = w * 4 + j;
                const unsigned short* bp = Whs + ((((nt * 16 + kt) * 32) + lane) << 4);
                v16bf bm = __builtin_bit_cast(v16bf, *(const v16us*)bp);
                acc[j] = __builtin_amdgcn_wmma_f32_16x16x32_bf16(
                             false, a, false, bm, (short)0, acc[j], false, false);
            }
        }

        // prefetch next step's xw while this step finishes
        if (t + 1 < TT) {
            issue_stage(t + 1, xbuf[1 - cur]);
            wait_async_le8();           // retire xw(t); xw(t+1) stays in flight
        } else {
            wait_async_0();
        }
        __syncthreads();                // hbuf reads done + xbuf[cur] visible

        // h_t = tanh(xw_t + hW); write f32 to out (in place of xw) + bf16 to LDS
        #pragma unroll
        for (int j = 0; j < 4; ++j) {
            int col = (w * 4 + j) * 16 + (lane & 15);
            #pragma unroll
            for (int r = 0; r < 8; ++r) {
                int row = mm + r;
                float v = fast_tanh(acc[j][r] + xbuf[cur][row * 516 + col]);
                out[((long)(n0 + row) * TT + t) * HH + col] = v;
                hbuf[row * 520 + col] = f2bfu(v);
            }
        }
        __syncthreads();                // hbuf ready for next step's GEMM
    }
}

// -----------------------------------------------------------------------------
// Launch: swizzle Wx,Wh (needs 1 MB of d_ws) -> big GEMM into d_out -> scan.
// -----------------------------------------------------------------------------
extern "C" void kernel_launch(void* const* d_in, const int* in_sizes, int n_in,
                              void* d_out, int out_size, void* d_ws, size_t ws_size,
                              hipStream_t stream) {
    const float* x  = (const float*)d_in[0];   // (N,T,D)
    const float* h0 = (const float*)d_in[1];   // (N,H)
    const float* Wx = (const float*)d_in[2];   // (D,H)
    const float* Wh = (const float*)d_in[3];   // (H,H)
    const float* b  = (const float*)d_in[4];   // (H,)
    float* out = (float*)d_out;                // (N,T,H)

    unsigned short* Wxs = (unsigned short*)d_ws;       // 512 KB
    unsigned short* Whs = Wxs + 512 * 512;             // 512 KB

    swizzle_w<<<dim3(1024), 256, 0, stream>>>(Wx, Wxs);
    swizzle_w<<<dim3(1024), 256, 0, stream>>>(Wh, Whs);
    gemm_xw<<<dim3(256, 8), 256, 0, stream>>>(x, Wxs, b, out);
    rnn_scan<<<dim3(4), 256, 0, stream>>>(h0, Whs, out);
}
